// MultiheadAttention_12146167513267
// MI455X (gfx1250) — matrix operand order repaired and checked by reading, not compiled, not run on hardware
//
#include <hip/hip_runtime.h>

typedef __bf16 bf16;
typedef bf16  v8bf  __attribute__((ext_vector_type(8)));
typedef bf16  v16bf __attribute__((ext_vector_type(16)));
typedef float v8f   __attribute__((ext_vector_type(8)));
typedef int   v4i   __attribute__((ext_vector_type(4)));

typedef v4i __attribute__((address_space(1)))* gptr4;   // global int4*
typedef v4i __attribute__((address_space(3)))* lptr4;   // LDS int4*

#define BB 2
#define SS 4096
#define DD 1024
#define HH 16
#define HD 64
#define NQKV 3072           /* 3*D */
#define ROWS (BB*SS)        /* 8192 */

#if __has_builtin(__builtin_amdgcn_global_load_async_to_lds_b128) && \
    __has_builtin(__builtin_amdgcn_s_wait_asynccnt)
#define USE_ASYNC_LDS 1
#else
#define USE_ASYNC_LDS 0
#endif

__device__ __forceinline__ v16bf join16(v8bf a, v8bf b) {
  return __builtin_shufflevector(a, b, 0,1,2,3,4,5,6,7,8,9,10,11,12,13,14,15);
}
__device__ __forceinline__ v16bf ld16(const bf16* p) { return *(const v16bf*)p; }
__device__ __forceinline__ v8bf  ld8 (const bf16* p) { return *(const v8bf*)p; }

__device__ __forceinline__ v8f bwmma(v16bf a, v16bf b, v8f c) {
  return __builtin_amdgcn_wmma_f32_16x16x32_bf16(false, a, false, b,
                                                 (short)0, c, false, false);
}

// 16-byte global->LDS copy: async direct-to-LDS on CDNA5, fallback via VGPRs
__device__ __forceinline__ void copy16(const bf16* g, bf16* l) {
#if USE_ASYNC_LDS
  __builtin_amdgcn_global_load_async_to_lds_b128((gptr4)g, (lptr4)l, 0, 0);
#else
  *(v8bf*)l = *(const v8bf*)g;
#endif
}
__device__ __forceinline__ void async_commit_wait() {
#if USE_ASYNC_LDS
  __builtin_amdgcn_s_wait_asynccnt(0);
#endif
}

// ---------------------------------------------------------------- converts
__global__ void cvt_x(const float* __restrict__ x, bf16* __restrict__ xb, int n) {
  int i = (blockIdx.x * blockDim.x + threadIdx.x) * 4;
  if (i < n) {
    float4 f = *(const float4*)(x + i);
    xb[i]   = (bf16)f.x; xb[i+1] = (bf16)f.y;
    xb[i+2] = (bf16)f.z; xb[i+3] = (bf16)f.w;
  }
}

__global__ void cvt_transpose(const float* __restrict__ W, bf16* __restrict__ WT,
                              int rows, int cols) {
  int i = blockIdx.x * blockDim.x + threadIdx.x;
  if (i < rows * cols) {
    int r = i / cols, c = i % cols;
    WT[(size_t)c * rows + r] = (bf16)W[i];   // WT[N][K]
  }
}

// ---------------------------------------------------------------- QKV GEMM
__global__ void __launch_bounds__(256) qkv_gemm(
    const bf16* __restrict__ xb, const bf16* __restrict__ WqkvT,
    const float* __restrict__ bqkv,
    bf16* __restrict__ qb, bf16* __restrict__ kb, bf16* __restrict__ vtb) {
  const int wid = threadIdx.x >> 5;
  const int lane = threadIdx.x & 31;
  const int lm = lane & 15, hi = lane >> 4;
  const int gw = blockIdx.x * 8 + wid;        // 24576 waves
  const int rt = gw / 48, ct = gw % 48;

  const bf16* arow = xb + (size_t)(rt * 16 + lm) * DD;
  const bf16* brow[4];
#pragma unroll
  for (int nt = 0; nt < 4; ++nt)
    brow[nt] = WqkvT + (size_t)(ct * 64 + nt * 16 + lm) * DD;

  v8f acc[4] = {};
  for (int kk = 0; kk < DD; kk += 32) {
    v16bf A = join16(ld8(arow + kk + hi * 16), ld8(arow + kk + hi * 16 + 8));
#pragma unroll
    for (int nt = 0; nt < 4; ++nt)
      acc[nt] = bwmma(A, ld16(brow[nt] + kk + hi * 16), acc[nt]);
  }
#pragma unroll
  for (int nt = 0; nt < 4; ++nt) {
    int j = ct * 64 + nt * 16 + lm;           // global col in [0,3072)
    float bias = bqkv[j];
    int which = j >> 10;                       // 0=q 1=k 2=v
    int d = j & 1023;
    int h = d >> 6, e = d & 63;
#pragma unroll
    for (int r = 0; r < 8; ++r) {
      int orow = rt * 16 + r + hi * 8;         // row in [0,8192)
      int b = orow >> 12, s = orow & (SS - 1);
      int bh = b * HH + h;
      bf16 bv = (bf16)(acc[nt][r] + bias);
      if (which == 0)      qb [((size_t)bh * SS + s) * HD + e] = bv;
      else if (which == 1) kb [((size_t)bh * SS + s) * HD + e] = bv;
      else                 vtb[((size_t)bh * HD + e) * SS + s] = bv; // transposed
    }
  }
}

// ---------------------------------------------------------------- flash attention
// block = 8 waves, one (b,h), 128 q-rows; K/V tiles staged in double-buffered LDS
// via async direct-to-LDS copies shared by all 8 waves (8x less global traffic).
__global__ void __launch_bounds__(256) flash_attn(
    const bf16* __restrict__ qb, const bf16* __restrict__ kb,
    const bf16* __restrict__ vtb, const float* __restrict__ mask,
    bf16* __restrict__ ctx) {
  // padded rows (multiples of 16B) to spread LDS banks
  __shared__ __attribute__((aligned(16))) bf16 kbuf[2][32][72]; // 64 used / row
  __shared__ __attribute__((aligned(16))) bf16 vbuf[2][64][40]; // 32 used / row
  __shared__ __attribute__((aligned(16))) bf16 plds[8][16][40]; // 32 used / row

  const int tid = threadIdx.x;
  const int wid = tid >> 5;
  const int lane = tid & 31;
  const int lm = lane & 15, hi = lane >> 4;
  const int bh   = blockIdx.x >> 5;            // 1024 blocks: 32 (b,h) x 32 q-blocks
  const int qblk = blockIdx.x & 31;
  const int qt   = qblk * 8 + wid;             // this wave's 16-row q tile
  const int b    = bh >> 4;
  const size_t headbase = (size_t)bh * SS * HD;
  const bf16* kgbase = kb  + headbase;                  // [S][64]
  const bf16* vgbase = vtb + (size_t)bh * HD * SS;      // [64][S]

  // stage one 32-key tile: K 32x64 bf16 (4KB) + V^T 64x32 bf16 (4KB)
  auto stage = [&](int buf, int kt) {
    const int k0 = kt * 32;
    {   // K: thread -> (row = tid/8, 16B chunk = tid%8)
      int row = tid >> 3, ch = (tid & 7) * 8;
      copy16(kgbase + (size_t)(k0 + row) * HD + ch, &kbuf[buf][row][ch]);
    }
    {   // V^T: thread -> (e-row = tid/4, 16B chunk = tid%4)
      int e = tid >> 2, ch = (tid & 3) * 8;
      copy16(vgbase + (size_t)e * SS + k0 + ch, &vbuf[buf][e][ch]);
    }
  };

  // q tile as two A operands (K = e in [0,32) and [32,64))
  const bf16* qrow = qb + headbase + (size_t)(qt * 16 + lm) * HD;
  v16bf Aq[2];
#pragma unroll
  for (int c = 0; c < 2; ++c)
    Aq[c] = join16(ld8(qrow + c * 32 + hi * 16), ld8(qrow + c * 32 + hi * 16 + 8));

  float m[8], l[8];
  v8f O[4] = {};
#pragma unroll
  for (int r = 0; r < 8; ++r) { m[r] = -1e30f; l[r] = 0.f; }

  const float scale = 0.125f;                  // 1/sqrt(64)
  const float* maskbase = mask + ((size_t)b * SS + qt * 16) * SS;

  stage(0, 0);
  int buf = 0;
  for (int kt = 0; kt < SS / 32; ++kt) {
    const int k0 = kt * 32;
    async_commit_wait();                       // our async copies landed
    __syncthreads();                           // everyone's copies landed
    if (kt + 1 < SS / 32) stage(buf ^ 1, kt + 1);  // overlap next tile's DMA

    // scores: two 16x16 f32 tiles, B operands from LDS (ds_load_b128 pairs)
    v8f s0 = {}, s1 = {};
#pragma unroll
    for (int c = 0; c < 2; ++c) {
      v16bf B0 = *(const v16bf*)&kbuf[buf][lm][c * 32 + hi * 16];
      v16bf B1 = *(const v16bf*)&kbuf[buf][16 + lm][c * 32 + hi * 16];
      s0 = bwmma(Aq[c], B0, s0);
      s1 = bwmma(Aq[c], B1, s1);
    }
    // scale + additive mask (C layout: elem r -> row r+hi*8, col lm)
#pragma unroll
    for (int r = 0; r < 8; ++r) {
      const float* mrow = maskbase + (size_t)(r + hi * 8) * SS + k0;
      s0[r] = s0[r] * scale + mrow[lm];
      s1[r] = s1[r] * scale + mrow[16 + lm];
    }
    // online softmax (row reductions across 16-lane halves; wave32-safe)
#pragma unroll
    for (int r = 0; r < 8; ++r) {
      float a = fmaxf(s0[r], s1[r]);
      a = fmaxf(a, __shfl_xor(a, 1, 32));
      a = fmaxf(a, __shfl_xor(a, 2, 32));
      a = fmaxf(a, __shfl_xor(a, 4, 32));
      a = fmaxf(a, __shfl_xor(a, 8, 32));
      float mn = fmaxf(m[r], a);
      float p0 = __expf(s0[r] - mn);
      float p1 = __expf(s1[r] - mn);
      float rs = p0 + p1;
      rs += __shfl_xor(rs, 1, 32);
      rs += __shfl_xor(rs, 2, 32);
      rs += __shfl_xor(rs, 4, 32);
      rs += __shfl_xor(rs, 8, 32);
      float alpha = __expf(m[r] - mn);
      l[r] = l[r] * alpha + rs;
      m[r] = mn;
#pragma unroll
      for (int nt = 0; nt < 4; ++nt) O[nt][r] *= alpha;
      int qr = r + hi * 8;
      plds[wid][qr][lm]      = (bf16)p0;       // C layout -> row-major LDS
      plds[wid][qr][16 + lm] = (bf16)p1;
    }
    // reload P in A layout (same-wave DS ops are in-order; no barrier needed)
    const bf16* prow = &plds[wid][lm][0];
    v16bf Ap = join16(*(const v8bf*)(prow + hi * 16),
                      *(const v8bf*)(prow + hi * 16 + 8));
    // O += P(16x32) * V(32x64); V^T rows from LDS
#pragma unroll
    for (int nt = 0; nt < 4; ++nt) {
      v16bf Bv = *(const v16bf*)&vbuf[buf][nt * 16 + lm][hi * 16];
      O[nt] = bwmma(Ap, Bv, O[nt]);
    }
    __syncthreads();                           // all waves done with buf
    buf ^= 1;
  }
  // normalize + store ctx[b][h][sp][e] (flat == reference's head-interleaved reshape)
#pragma unroll
  for (int r = 0; r < 8; ++r) {
    float inv = 1.0f / l[r];
    int sp = qt * 16 + r + hi * 8;
#pragma unroll
    for (int nt = 0; nt < 4; ++nt)
      ctx[headbase + (size_t)sp * HD + nt * 16 + lm] = (bf16)(O[nt][r] * inv);
  }
}

// ---------------------------------------------------------------- output GEMM
__global__ void __launch_bounds__(256) out_gemm(
    const bf16* __restrict__ ctx, const bf16* __restrict__ WoT,
    const float* __restrict__ bo, float* __restrict__ out) {
  const int wid = threadIdx.x >> 5;
  const int lane = threadIdx.x & 31;
  const int lm = lane & 15, hi = lane >> 4;
  const int gw = blockIdx.x * 8 + wid;         // 8192 waves
  const int rt = gw >> 4, ct = gw & 15;

  const bf16* arow = ctx + (size_t)(rt * 16 + lm) * DD;
  v8f acc[4] = {};
  for (int kk = 0; kk < DD; kk += 32) {
    v16bf A = join16(ld8(arow + kk + hi * 16), ld8(arow + kk + hi * 16 + 8));
#pragma unroll
    for (int nt = 0; nt < 4; ++nt)
      acc[nt] = bwmma(A, ld16(WoT + (size_t)(ct * 64 + nt * 16 + lm) * DD +
                              kk + hi * 16), acc[nt]);
  }
#pragma unroll
  for (int nt = 0; nt < 4; ++nt) {
    int j = ct * 64 + nt * 16 + lm;
    float bias = bo[j];
#pragma unroll
    for (int r = 0; r < 8; ++r) {
      int row = rt * 16 + r + hi * 8;
      out[(size_t)row * DD + j] = acc[nt][r] + bias;
    }
  }
}

// ---------------------------------------------------------------- launch
extern "C" void kernel_launch(void* const* d_in, const int* in_sizes, int n_in,
                              void* d_out, int out_size, void* d_ws, size_t ws_size,
                              hipStream_t stream) {
  (void)in_sizes; (void)n_in; (void)out_size; (void)ws_size;
  const float* x    = (const float*)d_in[0];
  const float* mask = (const float*)d_in[1];
  const float* Wqkv = (const float*)d_in[2];
  const float* bqkv = (const float*)d_in[3];
  const float* Wo   = (const float*)d_in[4];
  const float* bo   = (const float*)d_in[5];
  float* out = (float*)d_out;

  char* ws = (char*)d_ws;                       // 88 MB total
  bf16* xb    = (bf16*)(ws);                    // 16 MB  x as bf16
  bf16* WqkvT = (bf16*)(ws + (16u << 20));      //  6 MB  Wqkv^T bf16 [3072][1024]
  bf16* WoT   = (bf16*)(ws + (22u << 20));      //  2 MB  Wo^T bf16 [1024][1024]
  bf16* qb    = (bf16*)(ws + (24u << 20));      // 16 MB  q [B,H,S,64]
  bf16* kb    = (bf16*)(ws + (40u << 20));      // 16 MB  k [B,H,S,64]
  bf16* vtb   = (bf16*)(ws + (56u << 20));      // 16 MB  v^T [B,H,64,S]
  bf16* ctx   = (bf16*)(ws + (72u << 20));      // 16 MB  attn output (flat)

  cvt_x<<<(ROWS * DD) / (256 * 4), 256, 0, stream>>>(x, xb, ROWS * DD);
  cvt_transpose<<<(DD * NQKV + 255) / 256, 256, 0, stream>>>(Wqkv, WqkvT, DD, NQKV);
  cvt_transpose<<<(DD * DD + 255) / 256, 256, 0, stream>>>(Wo, WoT, DD, DD);

  qkv_gemm<<<(512 * 48) / 8, 256, 0, stream>>>(xb, WqkvT, bqkv, qb, kb, vtb);
  flash_attn<<<BB * HH * (SS / 128), 256, 0, stream>>>(qb, kb, vtb, mask, ctx);
  out_gemm<<<(512 * 16) / 8, 256, 0, stream>>>(ctx, WoT, bo, out);
}
